// ConvAttention_51943334477847
// MI455X (gfx1250) — matrix operand order repaired and checked by reading, not compiled, not run on hardware
//
#include <hip/hip_runtime.h>
#include <hip/hip_bf16.h>

// MI455X / gfx1250, wave32. All matmuls via v_wmma_f32_16x16x32_bf16.

typedef __attribute__((ext_vector_type(16))) __bf16 v16bf;
typedef __attribute__((ext_vector_type(8)))  __bf16 v8bf;
typedef __attribute__((ext_vector_type(8)))  float  v8f;
typedef __attribute__((ext_vector_type(4)))  float  v4f;

#define NB    8
#define CH    512
#define LQ    2048
#define NHEAD 8
#define DH    64

static __device__ __forceinline__ v16bf cat8(v8bf lo, v8bf hi) {
  return __builtin_shufflevector(lo, hi, 0,1,2,3,4,5,6,7,8,9,10,11,12,13,14,15);
}
static __device__ __forceinline__ v8f wmma_bf16(v16bf a, v16bf b, v8f c) {
  // (neg_a, A, neg_b, B, c_mod, C, reuse_a, reuse_b)
  return __builtin_amdgcn_wmma_f32_16x16x32_bf16(false, a, false, b, (short)0, c, false, false);
}
static __device__ __forceinline__ v8f v8f_zero() {
  v8f z = {0.f,0.f,0.f,0.f,0.f,0.f,0.f,0.f};
  return z;
}

// ---------------------------------------------------------------------------
// (1) weights fp32 -> bf16
__global__ void convert_w_kernel(const float* __restrict__ wq, const float* __restrict__ wk,
                                 const float* __restrict__ wv, const float* __restrict__ wo,
                                 __bf16* __restrict__ bwq, __bf16* __restrict__ bwk,
                                 __bf16* __restrict__ bwv, __bf16* __restrict__ bwo) {
  int i = blockIdx.x * blockDim.x + threadIdx.x;
  if (i < CH * CH) {
    bwq[i] = (__bf16)wq[i];
    bwk[i] = (__bf16)wk[i];
    bwv[i] = (__bf16)wv[i];
    bwo[i] = (__bf16)wo[i];
  }
}

// ---------------------------------------------------------------------------
// (2) x [N][C][L] fp32 -> xT [N][L][C] bf16 (LDS-tiled transpose)
__global__ void transpose_x_kernel(const float* __restrict__ x, __bf16* __restrict__ xT) {
  __shared__ float tile[32][33];
  int n  = blockIdx.z;
  int c0 = blockIdx.y * 32;
  int l0 = blockIdx.x * 32;
  int tx = threadIdx.x;            // 0..31 (along L on read, along C on write)
  int ty = threadIdx.y;            // 0..7
  for (int i = 0; i < 32; i += 8)
    tile[ty + i][tx] = x[((size_t)(n * CH + c0 + ty + i)) * LQ + l0 + tx];
  __syncthreads();
  for (int i = 0; i < 32; i += 8)
    xT[((size_t)(n * LQ + l0 + ty + i)) * CH + c0 + tx] = (__bf16)tile[tx][ty + i];
}

// ---------------------------------------------------------------------------
// (3) QKV projection: D(o,l) = sum_c W[o,c] * x[n,c,l]
//     A-frags = W rows, B-frags = xT rows (shared across Q/K/V).
//     Q,K stored [N][L][C] (row per position); V stored d-major [N][C][L].
__global__ void proj_qkv_kernel(const __bf16* __restrict__ xT,
                                const __bf16* __restrict__ bwq, const __bf16* __restrict__ bwk,
                                const __bf16* __restrict__ bwv,
                                __bf16* __restrict__ Q, __bf16* __restrict__ K,
                                __bf16* __restrict__ Vt) {
  int lane = threadIdx.x & 31;
  int wave = threadIdx.x >> 5;
  int n  = blockIdx.z;
  int o0 = blockIdx.y * 16;
  int l0 = blockIdx.x * 64 + wave * 16;
  int lm   = lane & 15;
  int ch8  = (lane < 16) ? 0 : 8;    // unused; A-frags load at ch16 too
  int ch16 = (lane < 16) ? 0 : 16;   // B-frag K-half base
  int hi8  = (lane < 16) ? 0 : 8;    // D-frag row-half base

  const __bf16* xrow  = xT  + ((size_t)(n * LQ + l0 + lm)) * CH;
  const __bf16* wqrow = bwq + (size_t)(o0 + lm) * CH;
  const __bf16* wkrow = bwk + (size_t)(o0 + lm) * CH;
  const __bf16* wvrow = bwv + (size_t)(o0 + lm) * CH;

  v8f dq = v8f_zero(), dk = v8f_zero(), dv = v8f_zero();
  for (int c0 = 0; c0 < CH; c0 += 32) {
    v16bf bx = *(const v16bf*)(xrow + c0 + ch16);
    v16bf aq = *(const v16bf*)(wqrow + c0 + ch16);
    v16bf ak = *(const v16bf*)(wkrow + c0 + ch16);
    v16bf av = *(const v16bf*)(wvrow + c0 + ch16);
    dq = wmma_bf16(aq, bx, dq);
    dk = wmma_bf16(ak, bx, dk);
    dv = wmma_bf16(av, bx, dv);
  }

  int l = l0 + lm;                     // this lane's column (position)
  v8bf q8, k8;
  for (int r = 0; r < 8; ++r) { q8[r] = (__bf16)dq[r]; k8[r] = (__bf16)dk[r]; }
  // rows o0+hi8+r are contiguous channels -> one 16B store each
  *(v8bf*)(Q + ((size_t)(n * LQ + l)) * CH + o0 + hi8) = q8;
  *(v8bf*)(K + ((size_t)(n * LQ + l)) * CH + o0 + hi8) = k8;
  for (int r = 0; r < 8; ++r)
    Vt[((size_t)(n * CH + o0 + hi8 + r)) * LQ + l] = (__bf16)dv[r];
}

// ---------------------------------------------------------------------------
// (4) causal flash attention, one 16-query tile per wave, 32 keys per step
__global__ void attention_kernel(const __bf16* __restrict__ Q, const __bf16* __restrict__ K,
                                 const __bf16* __restrict__ Vt, __bf16* __restrict__ Ao) {
  __shared__ __attribute__((aligned(32))) __bf16 pbuf[4][16 * 32];
  int lane = threadIdx.x & 31;
  int wave = threadIdx.x >> 5;
  int h = blockIdx.y, n = blockIdx.z;
  int q0 = (blockIdx.x * 4 + wave) * 16;
  int lm   = lane & 15;
  int ch8  = (lane < 16) ? 0 : 8;
  int ch16 = (lane < 16) ? 0 : 16;
  int hi8  = (lane < 16) ? 0 : 8;
  const float scale = 0.044194173824159216f;   // 512^-0.5 (full-C scale as in ref)
  const float NEG = -3.0e38f;

  // Q A-fragments for this query tile (d=0..31 and d=32..63)
  const __bf16* qrow = Q + ((size_t)(n * LQ + q0 + lm)) * CH + h * DH;
  v16bf qa0 = *(const v16bf*)(qrow + ch16);
  v16bf qa1 = *(const v16bf*)(qrow + 32 + ch16);

  float m[8], lsum[8];
  v8f oacc[4];
  for (int r = 0; r < 8; ++r) { m[r] = NEG; lsum[r] = 0.f; }
  for (int j = 0; j < 4; ++j) oacc[j] = v8f_zero();

  __bf16* pb = &pbuf[wave][0];
  const __bf16* Kbase = K  + ((size_t)n * LQ) * CH + h * DH;
  const __bf16* Vbase = Vt + ((size_t)(n * CH + h * DH)) * LQ;

  int nkb = (q0 + 16 + 31) >> 5;              // causal: key blocks up to the diagonal
  for (int kb = 0; kb < nkb; ++kb) {
    int kbase = kb * 32;
    // --- S = Q K^T for 32 keys (two 16-col fragments, K-dim = d split 32+32)
    const __bf16* k0p = Kbase + ((size_t)(kbase + lm)) * CH;
    const __bf16* k1p = Kbase + ((size_t)(kbase + 16 + lm)) * CH;
    v16bf b00 = *(const v16bf*)(k0p + ch16);
    v16bf b01 = *(const v16bf*)(k0p + 32 + ch16);
    v16bf b10 = *(const v16bf*)(k1p + ch16);
    v16bf b11 = *(const v16bf*)(k1p + 32 + ch16);
    v8f s0 = v8f_zero(), s1 = v8f_zero();
    s0 = wmma_bf16(qa0, b00, s0);
    s0 = wmma_bf16(qa1, b01, s0);
    s1 = wmma_bf16(qa0, b10, s1);
    s1 = wmma_bf16(qa1, b11, s1);

    if (kb + 1 < nkb)   // pull next K block toward L2/L0
      __builtin_prefetch(Kbase + ((size_t)(kbase + 32 + lm)) * CH, 0, 0);

    // --- online softmax per row (rows live across 16-lane half-waves)
    for (int r = 0; r < 8; ++r) {
      int row = q0 + hi8 + r;
      float a0 = s0[r] * scale;
      float a1 = s1[r] * scale;
      if (kbase + lm      > row) a0 = NEG;     // causal mask
      if (kbase + 16 + lm > row) a1 = NEG;
      float mx = fmaxf(a0, a1);
      mx = fmaxf(mx, __shfl_xor(mx, 1));
      mx = fmaxf(mx, __shfl_xor(mx, 2));
      mx = fmaxf(mx, __shfl_xor(mx, 4));
      mx = fmaxf(mx, __shfl_xor(mx, 8));
      float mnew = fmaxf(m[r], mx);
      float corr = __expf(m[r] - mnew);
      float p0 = __expf(a0 - mnew);
      float p1 = __expf(a1 - mnew);
      float rs = p0 + p1;
      rs += __shfl_xor(rs, 1);
      rs += __shfl_xor(rs, 2);
      rs += __shfl_xor(rs, 4);
      rs += __shfl_xor(rs, 8);
      lsum[r] = lsum[r] * corr + rs;
      m[r] = mnew;
      oacc[0][r] *= corr; oacc[1][r] *= corr; oacc[2][r] *= corr; oacc[3][r] *= corr;
      // P tile to LDS (row-major 16x32) for D-layout -> A-layout transpose
      pb[(hi8 + r) * 32 + lm]      = (__bf16)p0;
      pb[(hi8 + r) * 32 + 16 + lm] = (__bf16)p1;
    }

    // --- P A-fragment from LDS (LDS is in-order within a wave)
    const __bf16* prow = pb + lm * 32;
    v16bf pa = *(const v16bf*)(prow + ch16);

    // --- O += P V  (4 d-column fragments of 16)
    for (int j = 0; j < 4; ++j) {
      const __bf16* vp = Vbase + ((size_t)(j * 16 + lm)) * LQ + kbase + ch16;
      v16bf vb = *(const v16bf*)vp;
      oacc[j] = wmma_bf16(pa, vb, oacc[j]);
    }
  }

  // --- finalize: divide by row sums, store AttnOut [N][L][C] bf16
  __bf16* obase = Ao + ((size_t)(n * LQ)) * CH + h * DH;
  for (int r = 0; r < 8; ++r) {
    float inv = 1.0f / lsum[r];
    size_t l = (size_t)(q0 + hi8 + r);
    for (int j = 0; j < 4; ++j)
      obase[l * CH + j * 16 + lm] = (__bf16)(oacc[j][r] * inv);
  }
}

// ---------------------------------------------------------------------------
// (5) y[n,o,l] = sum_c Wo[o,c] * Ao[n,l,c] + bo[o]   (M=l tile, N=o tile)
//     A-frags = Ao rows (contiguous), B-frags = Wo rows (contiguous, no transpose).
__global__ void final_proj_kernel(const __bf16* __restrict__ Ao, const __bf16* __restrict__ bwo,
                                  const float* __restrict__ bo, float* __restrict__ y) {
  int lane = threadIdx.x & 31;
  int wave = threadIdx.x >> 5;
  int n  = blockIdx.z;
  int o0 = blockIdx.y * 16;
  int l0 = blockIdx.x * 64 + wave * 16;
  int lm   = lane & 15;
  int ch8  = (lane < 16) ? 0 : 8;
  int ch16 = (lane < 16) ? 0 : 16;
  int hi8  = (lane < 16) ? 0 : 8;

  const __bf16* arow = Ao  + ((size_t)(n * LQ + l0 + lm)) * CH;
  const __bf16* brow = bwo + (size_t)(o0 + lm) * CH;
  v8f acc = v8f_zero();
  for (int c0 = 0; c0 < CH; c0 += 32) {
    v16bf a = *(const v16bf*)(arow + c0 + ch16);
    v16bf b = *(const v16bf*)(brow + c0 + ch16);
    acc = wmma_bf16(a, b, acc);
  }
  int o = o0 + lm;                       // this lane's output channel (column)
  float bias = bo[o];
  v4f lo4 = { acc[0] + bias, acc[1] + bias, acc[2] + bias, acc[3] + bias };
  v4f hi4 = { acc[4] + bias, acc[5] + bias, acc[6] + bias, acc[7] + bias };
  // rows l0+hi8+r are contiguous positions -> two 16B stores
  float* yb = y + ((size_t)(n * CH + o)) * LQ + l0 + hi8;
  *(v4f*)(yb)     = lo4;
  *(v4f*)(yb + 4) = hi4;
}

// ---------------------------------------------------------------------------
extern "C" void kernel_launch(void* const* d_in, const int* in_sizes, int n_in,
                              void* d_out, int out_size, void* d_ws, size_t ws_size,
                              hipStream_t stream) {
  (void)in_sizes; (void)n_in; (void)out_size; (void)ws_size;
  const float* x  = (const float*)d_in[0];
  const float* Wq = (const float*)d_in[1];
  const float* Wk = (const float*)d_in[2];
  const float* Wv = (const float*)d_in[3];
  const float* Wo = (const float*)d_in[4];
  const float* bo = (const float*)d_in[5];
  float* y = (float*)d_out;

  char* ws = (char*)d_ws;
  const size_t szX = (size_t)NB * LQ * CH * sizeof(__bf16);   // 16 MiB each
  const size_t szW = (size_t)CH * CH * sizeof(__bf16);        // 0.5 MiB each
  __bf16* xT  = (__bf16*)ws;  ws += szX;
  __bf16* bwq = (__bf16*)ws;  ws += szW;
  __bf16* bwk = (__bf16*)ws;  ws += szW;
  __bf16* bwv = (__bf16*)ws;  ws += szW;
  __bf16* bwo = (__bf16*)ws;  ws += szW;
  __bf16* Qb  = (__bf16*)ws;  ws += szX;
  __bf16* Kb  = (__bf16*)ws;  ws += szX;
  __bf16* Vtb = (__bf16*)ws;  ws += szX;
  __bf16* Aob = (__bf16*)ws;  ws += szX;

  convert_w_kernel<<<(CH * CH + 255) / 256, 256, 0, stream>>>(Wq, Wk, Wv, Wo, bwq, bwk, bwv, bwo);
  transpose_x_kernel<<<dim3(LQ / 32, CH / 32, NB), dim3(32, 8), 0, stream>>>(x, xT);
  proj_qkv_kernel<<<dim3(LQ / 64, CH / 16, NB), 128, 0, stream>>>(xT, bwq, bwk, bwv, Qb, Kb, Vtb);
  attention_kernel<<<dim3(LQ / 64, NHEAD, NB), 128, 0, stream>>>(Qb, Kb, Vtb, Aob);
  final_proj_kernel<<<dim3(LQ / 64, CH / 16, NB), 128, 0, stream>>>(Aob, bwo, bo, y);
}
